// Net_3152505995976
// MI455X (gfx1250) — matrix operand order repaired and checked by reading, not compiled, not run on hardware
//
#include <hip/hip_runtime.h>

// ---------------------------------------------------------------------------
// PCA-frame point-cloud net for MI455X (gfx1250), wave32 + WMMA f16->f32.
//   init -> knn -> geom(eigh+layer1) -> stats/finalize/fold -> gemm L2 ->
//   stats/fold -> gemm L3 -> stats -> f3d -> fused-gather DD2 (wmma) ->
//   finalize -> head gemm (wmma, atomic-max pool) -> single-block head tail.
// CDNA5 specifics: v_wmma_f32_16x16x32_f16 for all big GEMMs, A-tiles staged
// with global_load_async_to_lds_b128 (ASYNCcnt), vectorized LDS fragment
// loads (ds_load_b128), wave32 layouts per ISA 7.12.2. Workspace ~124 MB.
// ---------------------------------------------------------------------------

#define BB   8
#define NN   2048
#define KK   20
#define BN   (BB * NN)           // 16384
#define MROW (BB * NN * KK)      // 327680

typedef _Float16 f16;
typedef __attribute__((ext_vector_type(16))) _Float16 v16h;
typedef __attribute__((ext_vector_type(8)))  _Float16 v8h;
typedef __attribute__((ext_vector_type(8)))  float    v8f;

#define NEG_INF (-3.0e38f)

// -------------------------- WMMA fragment helpers --------------------------
// Join two aligned 8-half LDS chunks into one 16-half fragment register set.
__device__ __forceinline__ v16h ld2x8(const f16* p0, const f16* p1) {
    v8h a = *(const v8h*)p0;
    v8h b = *(const v8h*)p1;
    return __builtin_shufflevector(a, b, 0, 1, 2, 3, 4, 5, 6, 7,
                                   8, 9, 10, 11, 12, 13, 14, 15);
}

// A fragment: 16x32 f16 (M x K), ISA layout: lane&15 = row, half = lane>>4.
// Halves are the two contiguous runs [half*8, half*8+7] and [16+half*8, ...].
__device__ __forceinline__ v16h load_a_frag(const f16* At, int ldA, int m0, int k0, int lane) {
    int half = (lane >> 4) & 1;
    int m = lane & 15;
    const f16* row = At + (size_t)(m0 + m) * ldA + k0 + half * 8;
    return ld2x8(row, row + 16);
}

// B fragment: 32x16 f16 (K x N) with W stored TRANSPOSED in LDS (Wt[col][k]):
// lane&15 = col, two 8-K runs [k0+half*8, +7] and [k0+16+half*8, +7].
__device__ __forceinline__ v16h load_b_frag(const f16* Wt, int ldB, int k0, int n0, int lane) {
    int half = (lane >> 4) & 1;
    int n = lane & 15;
    const f16* col = Wt + (size_t)(n0 + n) * ldB + k0 + half * 8;
    return ld2x8(col, col + 16);
}

__device__ __forceinline__ v8f wmma_f16(v16h a, v16h b, v8f c) {
    return __builtin_amdgcn_wmma_f32_16x16x32_f16(false, a, false, b, (short)0, c, false, false);
}

__device__ __forceinline__ unsigned float_key(float f) {
    unsigned u = __float_as_uint(f);
    return (u & 0x80000000u) ? ~u : (u | 0x80000000u);
}
__device__ __forceinline__ float key_float(unsigned k) {
    unsigned u = (k & 0x80000000u) ? (k & 0x7FFFFFFFu) : ~k;
    return __uint_as_float(u);
}

// CDNA5 async copy: 16 bytes global -> LDS per lane, tracked by ASYNCcnt.
__device__ __forceinline__ void async_g2l_b128(unsigned lds_off, const void* gaddr) {
    unsigned long long g = (unsigned long long)(uintptr_t)gaddr;
    asm volatile("global_load_async_to_lds_b128 %0, %1, off"
                 :: "v"(lds_off), "v"(g) : "memory");
}
__device__ __forceinline__ void async_wait0() {
    asm volatile("s_wait_asynccnt 0x0" ::: "memory");
}

// ------------------------------- init ---------------------------------------
__global__ void init_kernel(float* stats, unsigned* y1pool) {
    int t = blockIdx.x * blockDim.x + threadIdx.x;
    if (t < 640) stats[t] = 0.f;
    if (t < BB * 1024) y1pool[t] = 0u;   // below every real float's key
}

// ------------------------------- kNN ----------------------------------------
__global__ void knn_kernel(const float* __restrict__ pos, int* __restrict__ idx) {
    __shared__ float Px[NN], Py[NN], Pz[NN];
    int b = blockIdx.x;
    int i = blockIdx.y * 256 + threadIdx.x;
    const float* base = pos + (size_t)b * NN * 3;
    for (int t = threadIdx.x; t < NN; t += 256) {
        Px[t] = base[t * 3 + 0];
        Py[t] = base[t * 3 + 1];
        Pz[t] = base[t * 3 + 2];
    }
    __syncthreads();
    float qx = Px[i], qy = Py[i], qz = Pz[i];
    float bd[KK]; int bi[KK];
#pragma unroll
    for (int t = 0; t < KK; ++t) { bd[t] = 3.0e38f; bi[t] = 0; }
    for (int j = 0; j < NN; ++j) {
        float dx = Px[j] - qx, dy = Py[j] - qy, dz = Pz[j] - qz;
        float d = dx * dx + dy * dy + dz * dz;
        if (j == i) continue;
        if (d < bd[KK - 1]) {
            int pos_ = KK - 1;
#pragma unroll
            for (int t = KK - 1; t >= 1; --t) {
                if (bd[t - 1] > d) { bd[t] = bd[t - 1]; bi[t] = bi[t - 1]; pos_ = t - 1; }
            }
#pragma unroll
            for (int t = 0; t < KK; ++t)
                if (t == pos_) { bd[t] = d; bi[t] = j; }
        }
    }
    int* out = idx + ((size_t)b * NN + i) * KK;
#pragma unroll
    for (int t = 0; t < KK; ++t) out[t] = bi[t];
}

// --------------------------- 3x3 symmetric eigh -----------------------------
__device__ __forceinline__ void cross3(const float* a, const float* b, float* c) {
    c[0] = a[1] * b[2] - a[2] * b[1];
    c[1] = a[2] * b[0] - a[0] * b[2];
    c[2] = a[0] * b[1] - a[1] * b[0];
}

__device__ void eigvec3(float a00, float a01, float a02, float a11, float a12, float a22,
                        float lam, float v[3]) {
    float r0[3] = {a00 - lam, a01, a02};
    float r1[3] = {a01, a11 - lam, a12};
    float r2[3] = {a02, a12, a22 - lam};
    float c0[3], c1[3], c2[3];
    cross3(r0, r1, c0); cross3(r1, r2, c1); cross3(r0, r2, c2);
    float n0 = c0[0]*c0[0] + c0[1]*c0[1] + c0[2]*c0[2];
    float n1 = c1[0]*c1[0] + c1[1]*c1[1] + c1[2]*c1[2];
    float n2 = c2[0]*c2[0] + c2[1]*c2[1] + c2[2]*c2[2];
    float bx = c0[0], by = c0[1], bz = c0[2], nb = n0;
    if (n1 > nb) { bx = c1[0]; by = c1[1]; bz = c1[2]; nb = n1; }
    if (n2 > nb) { bx = c2[0]; by = c2[1]; bz = c2[2]; nb = n2; }
    if (nb < 1e-30f) { v[0] = 1.f; v[1] = 0.f; v[2] = 0.f; return; }
    float inv = rsqrtf(nb);
    v[0] = bx * inv; v[1] = by * inv; v[2] = bz * inv;
}

// V row-major [d][e]; column e = eigenvector for ascending eigenvalue e.
__device__ void eigh3(float a00, float a01, float a02, float a11, float a12, float a22,
                      float V[9]) {
    float q = (a00 + a11 + a22) * (1.f / 3.f);
    float b00 = a00 - q, b11 = a11 - q, b22 = a22 - q;
    float p1 = a01 * a01 + a02 * a02 + a12 * a12;
    float p2 = b00 * b00 + b11 * b11 + b22 * b22 + 2.f * p1;
    float p = sqrtf(p2 * (1.f / 6.f));
    if (p < 1e-12f) {
        int o0 = 0, o1 = 1, o2 = 2;
        float d0 = a00, d1 = a11, d2 = a22, tf; int ti;
        if (d0 > d1) { tf = d0; d0 = d1; d1 = tf; ti = o0; o0 = o1; o1 = ti; }
        if (d1 > d2) { tf = d1; d1 = d2; d2 = tf; ti = o1; o1 = o2; o2 = ti; }
        if (d0 > d1) { tf = d0; d0 = d1; d1 = tf; ti = o0; o0 = o1; o1 = ti; }
#pragma unroll
        for (int t = 0; t < 9; ++t) V[t] = 0.f;
        V[o0 * 3 + 0] = 1.f; V[o1 * 3 + 1] = 1.f; V[o2 * 3 + 2] = 1.f;
        return;
    }
    float inv = 1.f / p;
    float c00 = b00 * inv, c01 = a01 * inv, c02 = a02 * inv;
    float c11 = b11 * inv, c12 = a12 * inv, c22 = b22 * inv;
    float detB = c00 * (c11 * c22 - c12 * c12)
               - c01 * (c01 * c22 - c12 * c02)
               + c02 * (c01 * c12 - c11 * c02);
    float r = fminf(1.f, fmaxf(-1.f, detB * 0.5f));
    float phi = acosf(r) * (1.f / 3.f);
    float lmax = q + 2.f * p * cosf(phi);
    float lmin = q + 2.f * p * cosf(phi + 2.0943951023931953f);
    float v0[3], v2[3], v1[3];
    eigvec3(a00, a01, a02, a11, a12, a22, lmin, v0);
    eigvec3(a00, a01, a02, a11, a12, a22, lmax, v2);
    cross3(v2, v0, v1);
    float nn = v1[0]*v1[0] + v1[1]*v1[1] + v1[2]*v1[2];
    float ninv = (nn > 1e-30f) ? rsqrtf(nn) : 0.f;
    v1[0] *= ninv; v1[1] *= ninv; v1[2] *= ninv;
    V[0] = v0[0]; V[3] = v0[1]; V[6] = v0[2];
    V[1] = v1[0]; V[4] = v1[1]; V[7] = v1[2];
    V[2] = v2[0]; V[5] = v2[1]; V[8] = v2[2];
}

// ----------------- geom: rel, cov, eigh, loc, DD1 layer1 --------------------
__global__ void geom_kernel(const float* __restrict__ pos, const int* __restrict__ idx,
                            const float* __restrict__ w11, const float* __restrict__ b11,
                            float* __restrict__ Vout, float* __restrict__ loc,
                            f16* __restrict__ h1) {
    int bn = blockIdx.x * 256 + threadIdx.x;
    if (bn >= BN) return;
    int b = bn >> 11;
    const float* q = pos + (size_t)bn * 3;
    float qx = q[0], qy = q[1], qz = q[2];
    const int* nbr = idx + (size_t)bn * KK;
    float cxx = 0, cxy = 0, cxz = 0, cyy = 0, cyz = 0, czz = 0;
    for (int k = 0; k < KK; ++k) {
        const float* pj = pos + ((size_t)b * NN + nbr[k]) * 3;
        float rx = pj[0] - qx, ry = pj[1] - qy, rz = pj[2] - qz;
        cxx += rx * rx; cxy += rx * ry; cxz += rx * rz;
        cyy += ry * ry; cyz += ry * rz; czz += rz * rz;
    }
    float V[9];
    eigh3(cxx, cxy, cxz, cyy, cyz, czz, V);
#pragma unroll
    for (int t = 0; t < 9; ++t) Vout[(size_t)bn * 9 + t] = V[t];
    for (int k = 0; k < KK; ++k) {
        const float* pj = pos + ((size_t)b * NN + nbr[k]) * 3;
        float rx = pj[0] - qx, ry = pj[1] - qy, rz = pj[2] - qz;
        float l0 = rx * V[0] + ry * V[3] + rz * V[6];
        float l1 = rx * V[1] + ry * V[4] + rz * V[7];
        float l2 = rx * V[2] + ry * V[5] + rz * V[8];
        size_t row = (size_t)bn * KK + k;
        loc[row * 3 + 0] = l0; loc[row * 3 + 1] = l1; loc[row * 3 + 2] = l2;
        f16* hout = h1 + row * 64;
        for (int o = 0; o < 64; ++o) {
            float h = b11[o] + l0 * w11[o] + l1 * w11[64 + o] + l2 * w11[128 + o];
            hout[o] = (f16)fmaxf(h, 0.f);
        }
    }
}

// --------------------------- BN stats / finalize / fold ---------------------
__global__ void bn_stats_kernel(const f16* __restrict__ X, int M, int C,
                                float* __restrict__ stats) {
    int c = threadIdx.x % C;
    int rg = threadIdx.x / C;
    int nrg = blockDim.x / C;
    float s = 0.f, ss = 0.f;
    for (size_t r = (size_t)blockIdx.x * nrg + rg; r < (size_t)M; r += (size_t)gridDim.x * nrg) {
        float v = (float)X[r * C + c];
        s += v; ss += v * v;
    }
    atomicAdd(&stats[c], s);
    atomicAdd(&stats[C + c], ss);
}

__global__ void bn_finalize_kernel(const float* __restrict__ stats,
                                   const float* __restrict__ g, const float* __restrict__ be,
                                   float cntInv, int C,
                                   float* __restrict__ scale, float* __restrict__ shift) {
    int c = threadIdx.x;
    if (c >= C) return;
    float m = stats[c] * cntInv;
    float v = stats[C + c] * cntInv - m * m;
    float s = g[c] * rsqrtf(v + 1e-5f);
    scale[c] = s;
    shift[c] = be[c] - m * s;
}

// Fold input-BN shift into next layer's bias: bias'[n] = bias[n] + sum_k shift[k]*W[k,n]
__global__ void fold_bias_kernel(const float* __restrict__ shift, const float* __restrict__ W,
                                 const float* __restrict__ bias, int K, int Nc,
                                 float* __restrict__ bias2) {
    int n = blockIdx.x * blockDim.x + threadIdx.x;
    if (n >= Nc) return;
    float acc = bias[n];
    for (int k = 0; k < K; ++k) acc += shift[k] * W[(size_t)k * Nc + n];
    bias2[n] = acc;
}

// -------------------- generic WMMA GEMM (relu store / atomic max) -----------
// MODE 0: out f16 = relu(A @ (wscale.W) + bias)    (raw, pre-BN)
// MODE 1: atomicMax ordered-key of (A @ W + bias) into [B,Nc] pool
// A is staged raw via CDNA5 async global->LDS; input-BN scale folded into W.
template <int MODE>
__global__ void gemm_kernel(const f16* __restrict__ A,
                            const float* __restrict__ wscale,
                            const float* __restrict__ W, const float* __restrict__ bias,
                            void* __restrict__ out, int M, int K, int Nc) {
    __shared__ f16 At[64][136];   // row-major A tile, K <= 128, 272B pitch (16B mult)
    __shared__ f16 Wt[64][136];   // TRANSPOSED W tile: Wt[col][k]
    int mBase = blockIdx.x * 64;
    int nBase = blockIdx.y * 64;
    int tid = threadIdx.x;        // 128 threads = 4 waves
    // --- async stage A (pure f16 copy, 16B per lane per op) ---
    {
        const char* Ag = (const char*)(A + (size_t)mBase * K);
        unsigned ldsBase = (unsigned)(uintptr_t)&At[0][0];
        int cpr = K >> 3;                       // 16B chunks per row
        for (int t = tid; t < 64 * cpr; t += 128) {
            int r = t / cpr, c = t - r * cpr;
            async_g2l_b128(ldsBase + (unsigned)(r * (136 * 2) + c * 16),
                           Ag + (size_t)r * K * 2 + c * 16);
        }
    }
    // --- stage W transposed (+ folded BN scale) ---
    for (int t = tid; t < K * 64; t += 128) {
        int k = t >> 6, c = t & 63;
        float w = W[(size_t)k * Nc + nBase + c];
        if (wscale) w *= wscale[k];
        Wt[c][k] = (f16)w;
    }
    async_wait0();
    __syncthreads();
    int wave = tid >> 5, lane = tid & 31;
    int mOff = wave * 16;
    v8f acc[4];
#pragma unroll
    for (int nt = 0; nt < 4; ++nt)
        acc[nt] = (v8f){0.f, 0.f, 0.f, 0.f, 0.f, 0.f, 0.f, 0.f};
    for (int k0 = 0; k0 < K; k0 += 32) {
        v16h a = load_a_frag(&At[0][0], 136, mOff, k0, lane);
#pragma unroll
        for (int nt = 0; nt < 4; ++nt) {
            v16h bfr = load_b_frag(&Wt[0][0], 136, k0, nt * 16, lane);
            acc[nt] = wmma_f16(a, bfr, acc[nt]);
        }
    }
    int half = lane >> 4, n = lane & 15;
#pragma unroll
    for (int nt = 0; nt < 4; ++nt) {
        int col = nBase + nt * 16 + n;
        float bb = bias[col];
#pragma unroll
        for (int r = 0; r < 8; ++r) {
            int m = mBase + mOff + r + half * 8;
            float v = acc[nt][r] + bb;
            if (MODE == 0) {
                ((f16*)out)[(size_t)m * Nc + col] = (f16)fmaxf(v, 0.f);
            } else {
                atomicMax(&((unsigned*)out)[(m >> 11) * Nc + col], float_key(v));
            }
        }
    }
}

// -------------- f3d: directional max pool of BN(h3) x loc -------------------
__global__ void f3d_kernel(const f16* __restrict__ h3,
                           const float* __restrict__ sc, const float* __restrict__ sh,
                           const float* __restrict__ loc, float* __restrict__ f3d) {
    int t = blockIdx.x * 256 + threadIdx.x;      // over BN*64
    int bn = t >> 6, c = t & 63;
    float s = sc[c], o = sh[c];
    float m0 = NEG_INF, m1 = NEG_INF, m2 = NEG_INF;
    size_t rbase = (size_t)bn * KK;
    for (int k = 0; k < KK; ++k) {
        float h = (float)h3[(rbase + k) * 64 + c] * s + o;
        const float* L = loc + (rbase + k) * 3;
        m0 = fmaxf(m0, h * L[0]);
        m1 = fmaxf(m1, h * L[1]);
        m2 = fmaxf(m2, h * L[2]);
    }
    f3d[(size_t)t * 3 + 0] = m0;
    f3d[(size_t)t * 3 + 1] = m1;
    f3d[(size_t)t * 3 + 2] = m2;
}

// ----- DD2: fused neighbor-gather + frame-rotate + WMMA GEMM + K pooling ----
// One block per point (bn); blockIdx.y selects the 64-column half of 128 outs.
__global__ void dd2_kernel(const float* __restrict__ f3d, const float* __restrict__ V,
                           const int* __restrict__ idx,
                           const float* __restrict__ W21, const float* __restrict__ b21,
                           float* __restrict__ pmax, float* __restrict__ pmin,
                           float* __restrict__ stats) {
    __shared__ f16 At[32][200];      // rows 20..31 zero-padded; 400B pitch (16B mult)
    __shared__ f16 Wt[64][200];      // TRANSPOSED: Wt[col][k], K=192
    __shared__ float h2s[32][65];
    __shared__ float Vs[9];
    __shared__ int js[KK];
    int bn = blockIdx.x;
    int colBase = blockIdx.y * 64;
    int tid = threadIdx.x;           // 128 threads = 4 waves
    int b = bn >> 11;
    if (tid < 9) Vs[tid] = V[(size_t)bn * 9 + tid];
    if (tid < KK) js[tid] = idx[(size_t)bn * KK + tid];
    for (int t = tid; t < 12 * 200; t += 128)
        At[20 + t / 200][t % 200] = (f16)0.f;
    __syncthreads();
    // stage A: row k, feature col c*3+e = dot(f3d[b, j_k, c, :], V[:, e])
    for (int t = tid; t < KK * 64; t += 128) {
        int k = t >> 6, c = t & 63;
        const float* f = f3d + (((size_t)b * NN + js[k]) * 64 + c) * 3;
        float f0 = f[0], f1 = f[1], f2 = f[2];
#pragma unroll
        for (int e = 0; e < 3; ++e)
            At[k][c * 3 + e] = (f16)(f0 * Vs[e] + f1 * Vs[3 + e] + f2 * Vs[6 + e]);
    }
    for (int t = tid; t < 192 * 64; t += 128) {
        int k = t >> 6, c = t & 63;
        Wt[c][k] = (f16)W21[(size_t)k * 128 + colBase + c];
    }
    __syncthreads();
    int wave = tid >> 5, lane = tid & 31;
    int nOff = wave * 16;            // each wave owns one 16-col tile
    v8f acc0 = (v8f){0.f, 0.f, 0.f, 0.f, 0.f, 0.f, 0.f, 0.f};
    v8f acc1 = acc0;
    for (int k0 = 0; k0 < 192; k0 += 32) {
        v16h bfr = load_b_frag(&Wt[0][0], 200, k0, nOff, lane);
        v16h a0 = load_a_frag(&At[0][0], 200, 0, k0, lane);
        acc0 = wmma_f16(a0, bfr, acc0);
        v16h a1 = load_a_frag(&At[0][0], 200, 16, k0, lane);
        acc1 = wmma_f16(a1, bfr, acc1);
    }
    int half = lane >> 4, n = lane & 15;
#pragma unroll
    for (int r = 0; r < 8; ++r) {
        h2s[r + half * 8][nOff + n] = acc0[r];
        h2s[16 + r + half * 8][nOff + n] = acc1[r];
    }
    __syncthreads();
    if (tid < 64) {
        int col = colBase + tid;
        float bb = b21[col];
        float mx = NEG_INF, mn = 3.0e38f, s = 0.f, ss = 0.f;
        for (int k = 0; k < KK; ++k) {
            float v = fmaxf(h2s[k][tid] + bb, 0.f);
            mx = fmaxf(mx, v); mn = fminf(mn, v);
            s += v; ss += v * v;
        }
        pmax[(size_t)bn * 128 + col] = mx;
        pmin[(size_t)bn * 128 + col] = mn;
        atomicAdd(&stats[col], s);
        atomicAdd(&stats[128 + col], ss);
    }
}

// fdd2 = BN(maxpool): scale>=0 -> scale*max+shift else scale*min+shift
__global__ void fdd2_fin_kernel(const float* __restrict__ pmax, const float* __restrict__ pmin,
                                const float* __restrict__ sc, const float* __restrict__ sh,
                                f16* __restrict__ ahead) {
    int t = blockIdx.x * 256 + threadIdx.x;      // over BN*128
    int c = t & 127;
    float s = sc[c];
    float v = (s >= 0.f) ? s * pmax[t] + sh[c] : s * pmin[t] + sh[c];
    ahead[t] = (f16)v;
}

// --------------------------- head tail (one block) --------------------------
__global__ void head_tail_kernel(const unsigned* __restrict__ y1pool,
                                 const float* __restrict__ gh1, const float* __restrict__ beh1,
                                 const float* __restrict__ wh2, const float* __restrict__ bh2,
                                 const float* __restrict__ gh2, const float* __restrict__ beh2,
                                 const float* __restrict__ wh3, const float* __restrict__ bh3,
                                 const float* __restrict__ gh3, const float* __restrict__ beh3,
                                 const float* __restrict__ wh4, const float* __restrict__ bh4,
                                 float* __restrict__ out) {
    __shared__ float s1[BB][1024];
    __shared__ float s2[BB][512];
    __shared__ float s3[BB][265];
    __shared__ float s4[BB][40];
    __shared__ float rowm[BB], rowl[BB];
    int t = threadIdx.x;             // 1024 threads
    {
        float v[BB]; float m = 0.f;
#pragma unroll
        for (int b = 0; b < BB; ++b) {
            v[b] = fmaxf(key_float(y1pool[b * 1024 + t]), 0.f);
            m += v[b];
        }
        m *= (1.f / BB);
        float var = 0.f;
#pragma unroll
        for (int b = 0; b < BB; ++b) { float d = v[b] - m; var += d * d; }
        var *= (1.f / BB);
        float sc = gh1[t] * rsqrtf(var + 1e-5f);
#pragma unroll
        for (int b = 0; b < BB; ++b) s1[b][t] = (v[b] - m) * sc + beh1[t];
    }
    __syncthreads();
#pragma unroll
    for (int p = 0; p < 4; ++p) {
        int o = t + p * 1024;        // 8*512 outputs
        int b = o >> 9, c = o & 511;
        float acc = bh2[c];
        for (int i = 0; i < 1024; ++i) acc += s1[b][i] * wh2[i * 512 + c];
        s2[b][c] = fmaxf(acc, 0.f);
    }
    __syncthreads();
    if (t < 512) {
        float m = 0.f;
#pragma unroll
        for (int b = 0; b < BB; ++b) m += s2[b][t];
        m *= (1.f / BB);
        float var = 0.f;
#pragma unroll
        for (int b = 0; b < BB; ++b) { float d = s2[b][t] - m; var += d * d; }
        var *= (1.f / BB);
        float sc = gh2[t] * rsqrtf(var + 1e-5f);
#pragma unroll
        for (int b = 0; b < BB; ++b) s2[b][t] = (s2[b][t] - m) * sc + beh2[t];
    }
    __syncthreads();
#pragma unroll
    for (int p = 0; p < 3; ++p) {
        int o = t + p * 1024;        // 8*265 outputs
        if (o < BB * 265) {
            int b = o / 265, c = o % 265;
            float acc = bh3[c];
            for (int i = 0; i < 512; ++i) acc += s2[b][i] * wh3[i * 265 + c];
            s3[b][c] = fmaxf(acc, 0.f);
        }
    }
    __syncthreads();
    if (t < 265) {
        float m = 0.f;
#pragma unroll
        for (int b = 0; b < BB; ++b) m += s3[b][t];
        m *= (1.f / BB);
        float var = 0.f;
#pragma unroll
        for (int b = 0; b < BB; ++b) { float d = s3[b][t] - m; var += d * d; }
        var *= (1.f / BB);
        float sc = gh3[t] * rsqrtf(var + 1e-5f);
#pragma unroll
        for (int b = 0; b < BB; ++b) s3[b][t] = (s3[b][t] - m) * sc + beh3[t];
    }
    __syncthreads();
    if (t < BB * 40) {
        int b = t / 40, c = t % 40;
        float acc = bh4[c];
        for (int i = 0; i < 265; ++i) acc += s3[b][i] * wh4[i * 40 + c];
        s4[b][c] = acc;
    }
    __syncthreads();
    if (t < BB) {
        float mx = NEG_INF;
        for (int c = 0; c < 40; ++c) mx = fmaxf(mx, s4[t][c]);
        float se = 0.f;
        for (int c = 0; c < 40; ++c) se += expf(s4[t][c] - mx);
        rowm[t] = mx; rowl[t] = logf(se);
    }
    __syncthreads();
    if (t < BB * 40) {
        int b = t / 40, c = t % 40;
        out[t] = s4[b][c] - rowm[b] - rowl[b];
    }
}

// ------------------------------- host side ----------------------------------
static inline size_t al256(size_t x) { return (x + 255) & ~(size_t)255; }

extern "C" void kernel_launch(void* const* d_in, const int* in_sizes, int n_in,
                              void* d_out, int out_size, void* d_ws, size_t ws_size,
                              hipStream_t stream) {
    const float* pos  = (const float*)d_in[0];
    const float* w11 = (const float*)d_in[1];  const float* b11 = (const float*)d_in[2];
    const float* g11 = (const float*)d_in[3];  const float* be11 = (const float*)d_in[4];
    const float* w12 = (const float*)d_in[5];  const float* b12 = (const float*)d_in[6];
    const float* g12 = (const float*)d_in[7];  const float* be12 = (const float*)d_in[8];
    const float* w13 = (const float*)d_in[9];  const float* b13 = (const float*)d_in[10];
    const float* g13 = (const float*)d_in[11]; const float* be13 = (const float*)d_in[12];
    const float* w21 = (const float*)d_in[13]; const float* b21 = (const float*)d_in[14];
    const float* g21 = (const float*)d_in[15]; const float* be21 = (const float*)d_in[16];
    const float* wh1 = (const float*)d_in[17]; const float* bh1 = (const float*)d_in[18];
    const float* gh1 = (const float*)d_in[19]; const float* beh1 = (const float*)d_in[20];
    const float* wh2 = (const float*)d_in[21]; const float* bh2 = (const float*)d_in[22];
    const float* gh2 = (const float*)d_in[23]; const float* beh2 = (const float*)d_in[24];
    const float* wh3 = (const float*)d_in[25]; const float* bh3 = (const float*)d_in[26];
    const float* gh3 = (const float*)d_in[27]; const float* beh3 = (const float*)d_in[28];
    const float* wh4 = (const float*)d_in[29]; const float* bh4 = (const float*)d_in[30];

    // ---- workspace layout (~124 MB) ----
    char* ws = (char*)d_ws;
    size_t off = 0;
    int*   idx   = (int*)(ws + off);   off = al256(off + (size_t)MROW * 4);
    float* Vbuf  = (float*)(ws + off); off = al256(off + (size_t)BN * 9 * 4);
    float* loc   = (float*)(ws + off); off = al256(off + (size_t)MROW * 3 * 4);
    f16*   h1    = (f16*)(ws + off);   off = al256(off + (size_t)MROW * 64 * 2);
    f16*   h2    = (f16*)(ws + off);   off = al256(off + (size_t)MROW * 64 * 2);
    float* f3d   = (float*)(ws + off); off = al256(off + (size_t)BN * 192 * 4);
    float* pmax  = (float*)(ws + off); off = al256(off + (size_t)BN * 128 * 4);
    float* pmin  = (float*)(ws + off); off = al256(off + (size_t)BN * 128 * 4);
    f16*   ahead = (f16*)(ws + off);   off = al256(off + (size_t)BN * 128 * 2);
    unsigned* y1pool = (unsigned*)(ws + off); off = al256(off + (size_t)BB * 1024 * 4);
    float* stats = (float*)(ws + off); off = al256(off + 640 * 4);  // l1,l2,l3 @0/128/256, dd2 @384
    float* ssbuf = (float*)(ws + off); off = al256(off + 640 * 4);  // sc/sh pairs
    float* fbias = (float*)(ws + off); off = al256(off + 128 * 4);  // folded biases l2,l3
    (void)ws_size; (void)n_in; (void)in_sizes; (void)out_size;

    float* st1 = stats;       float* st2 = stats + 128;
    float* st3 = stats + 256; float* stD = stats + 384;
    float* sc1 = ssbuf;       float* sh1 = ssbuf + 64;
    float* sc2 = ssbuf + 128; float* sh2 = ssbuf + 192;
    float* sc3 = ssbuf + 256; float* sh3 = ssbuf + 320;
    float* scD = ssbuf + 384; float* shD = ssbuf + 512;
    float* fb2 = fbias;       float* fb3 = fbias + 64;

    const float invM = 1.0f / (float)MROW;

    init_kernel<<<32, 256, 0, stream>>>(stats, y1pool);
    knn_kernel<<<dim3(BB, NN / 256), 256, 0, stream>>>(pos, idx);
    geom_kernel<<<BN / 256, 256, 0, stream>>>(pos, idx, w11, b11, Vbuf, loc, h1);

    bn_stats_kernel<<<256, 256, 0, stream>>>(h1, MROW, 64, st1);
    bn_finalize_kernel<<<1, 64, 0, stream>>>(st1, g11, be11, invM, 64, sc1, sh1);
    fold_bias_kernel<<<1, 64, 0, stream>>>(sh1, w12, b12, 64, 64, fb2);
    gemm_kernel<0><<<dim3(MROW / 64, 1), 128, 0, stream>>>(h1, sc1, w12, fb2,
                                                           (void*)h2, MROW, 64, 64);
    bn_stats_kernel<<<256, 256, 0, stream>>>(h2, MROW, 64, st2);
    bn_finalize_kernel<<<1, 64, 0, stream>>>(st2, g12, be12, invM, 64, sc2, sh2);
    fold_bias_kernel<<<1, 64, 0, stream>>>(sh2, w13, b13, 64, 64, fb3);
    gemm_kernel<0><<<dim3(MROW / 64, 1), 128, 0, stream>>>(h2, sc2, w13, fb3,
                                                           (void*)h1, MROW, 64, 64);
    bn_stats_kernel<<<256, 256, 0, stream>>>(h1, MROW, 64, st3);
    bn_finalize_kernel<<<1, 64, 0, stream>>>(st3, g13, be13, invM, 64, sc3, sh3);

    f3d_kernel<<<(BN * 64) / 256, 256, 0, stream>>>(h1, sc3, sh3, loc, f3d);
    dd2_kernel<<<dim3(BN, 2), 128, 0, stream>>>(f3d, Vbuf, idx, w21, b21, pmax, pmin, stD);
    bn_finalize_kernel<<<1, 128, 0, stream>>>(stD, g21, be21, invM, 128, scD, shD);
    fdd2_fin_kernel<<<(BN * 128) / 256, 256, 0, stream>>>(pmax, pmin, scD, shD, ahead);

    gemm_kernel<1><<<dim3(BN / 64, 1024 / 64), 128, 0, stream>>>(ahead, nullptr,
                                                                 wh1, bh1, (void*)y1pool,
                                                                 BN, 128, 1024);
    head_tail_kernel<<<1, 1024, 0, stream>>>(y1pool, gh1, beh1, wh2, bh2, gh2, beh2,
                                             wh3, bh3, gh3, beh3, wh4, bh4, (float*)d_out);
}
